// TSPTransformer_19207093748257
// MI455X (gfx1250) — compile-verified, matrix operand order checked by reading, not run on hardware
//
#include <hip/hip_runtime.h>

#define B_   64
#define S_   512
#define D_   512
#define H_   8
#define HD_  64
#define LYR  6
#define DFF_ 2048
#define M_   (B_ * S_)   // 32768 tokens

typedef __bf16 bf16;
typedef __attribute__((ext_vector_type(16))) __bf16 v16bf;
typedef __attribute__((ext_vector_type(8)))  __bf16 v8bf;
typedef __attribute__((ext_vector_type(4)))  __bf16 v4bf;
typedef __attribute__((ext_vector_type(8)))  float  v8f;
typedef __attribute__((ext_vector_type(4)))  float  v4f;
typedef __attribute__((ext_vector_type(4)))  int    v4i;

#ifndef __has_builtin
#define __has_builtin(x) 0
#endif

#if __has_builtin(__builtin_amdgcn_global_load_async_to_lds_b128)
#define HAVE_ASYNC_LDS 1
#else
#define HAVE_ASYNC_LDS 0
#endif

// Copy 16 bytes global -> LDS. Prefer the CDNA5 async path (ASYNCcnt-tracked
// GLOBAL_LOAD_ASYNC_TO_LDS_B128); fall back to load+ds_store.
static __device__ __forceinline__ void async_copy16(void* lds_dst,
                                                    const void* gsrc) {
#if HAVE_ASYNC_LDS
  __builtin_amdgcn_global_load_async_to_lds_b128(
      (__attribute__((address_space(1))) v4i*)gsrc,
      (__attribute__((address_space(3))) v4i*)lds_dst, 0, 0);
#else
  *(v8bf*)lds_dst = *(const v8bf*)gsrc;
#endif
}

static __device__ __forceinline__ void async_wait0() {
#if HAVE_ASYNC_LDS
#if __has_builtin(__builtin_amdgcn_s_wait_asynccnt)
  __builtin_amdgcn_s_wait_asynccnt(0);
#else
  asm volatile("s_wait_asynccnt 0x0" ::: "memory");
#endif
#endif
}

// ---------------------------------------------------------------------------
// WMMA helper: D(f32 16x16) = A(bf16 16x32) * B(bf16 32x16) + C
// ---------------------------------------------------------------------------
static __device__ __forceinline__ v8f wmma_bf16(v16bf a, v16bf b, v8f c) {
  return __builtin_amdgcn_wmma_f32_16x16x32_bf16(false, a, false, b,
                                                 (short)0, c, false, false);
}

// Fragment loader for a row-major [rows x ld] bf16 matrix per the CDNA5
// 16-bit A/B layout: lanes 0-15 hold K{k0..+7, k0+16..+23}, lanes 16-31 hold
// K{k0+8..+15, k0+24..+31}. Two 16-byte loads per lane.
static __device__ __forceinline__ v16bf load_frag(const bf16* base, int ld,
                                                  int row0, int k0, int lane) {
  int row = row0 + (lane & 15);
  int kh  = (lane >> 4) << 3;   // 0 or 8
  const bf16* p = base + (size_t)row * ld + k0 + kh;
  v16bf f;
  *(v8bf*)&f       = *(const v8bf*)(p);
  *((v8bf*)&f + 1) = *(const v8bf*)(p + 16);
  return f;
}

// ---------------------------------------------------------------------------
// f32 -> bf16 elementwise convert (grid-stride, 4 elems/thread)
// ---------------------------------------------------------------------------
__global__ __launch_bounds__(256) void k_f32_to_bf16(
    const float* __restrict__ in, bf16* __restrict__ out, long n) {
  long i      = ((long)blockIdx.x * blockDim.x + threadIdx.x) * 4;
  long stride = (long)gridDim.x * blockDim.x * 4;
  for (; i < n; i += stride) {
    v4f x = *(const v4f*)(in + i);
    v4bf y;
    y[0] = (bf16)x[0]; y[1] = (bf16)x[1];
    y[2] = (bf16)x[2]; y[3] = (bf16)x[3];
    *(v4bf*)(out + i) = y;
  }
}

// ---------------------------------------------------------------------------
// GEMM: C[M,N] = A[M,K](bf16) @ W[N,K]^T(bf16) + bias, optional ReLU.
// Block = 256 thr (8 waves). Block tile 128(M) x 256(N), K-step 32,
// double-buffered async global->LDS staging. Wave tile 64x64 = 4x4 fragments
// -> 16 WMMAs per K-step vs 16 ds_load_b128 (1 load per WMMA).
// ---------------------------------------------------------------------------
template <int OUT_BF16, int RELU>
__global__ __launch_bounds__(256) void k_gemm(
    const bf16* __restrict__ A, const bf16* __restrict__ W,
    const float* __restrict__ bias, void* __restrict__ Cout,
    int M, int N, int K) {
  __shared__ __align__(16) bf16 As[2][128][40];   // 128x32 (+pad)
  __shared__ __align__(16) bf16 Bs[2][256][40];   // 256x32 (+pad)

  const int lane = threadIdx.x & 31;
  const int wid  = threadIdx.x >> 5;
  const int wm   = (wid & 1) * 64;                // wave M offset in tile
  const int wn   = (wid >> 1) * 64;               // wave N offset in tile
  const int mblk = blockIdx.y * 128;
  const int nblk = blockIdx.x * 256;

  auto stage = [&](int buf, int kk) {
#pragma unroll
    for (int r = 0; r < 2; ++r) {   // A tile: 128 rows x 32 cols = 512 segs
      const int s   = threadIdx.x + 256 * r;
      const int row = s >> 2, c8 = (s & 3) * 8;
      async_copy16(&As[buf][row][c8], A + (size_t)(mblk + row) * K + kk + c8);
    }
#pragma unroll
    for (int r = 0; r < 4; ++r) {   // B tile: 256 rows x 32 cols = 1024 segs
      const int s   = threadIdx.x + 256 * r;
      const int row = s >> 2, c8 = (s & 3) * 8;
      async_copy16(&Bs[buf][row][c8], W + (size_t)(nblk + row) * K + kk + c8);
    }
  };

  v8f acc[4][4];
#pragma unroll
  for (int i = 0; i < 4; ++i)
#pragma unroll
    for (int j = 0; j < 4; ++j) acc[i][j] = (v8f){};

  stage(0, 0);
  const int nk = K >> 5;
  for (int t = 0; t < nk; ++t) {
    async_wait0();
    __syncthreads();                       // staged tile visible to all waves
    if (t + 1 < nk) stage((t + 1) & 1, (t + 1) << 5);
    const int buf = t & 1;

    v16bf afr[4];
#pragma unroll
    for (int i = 0; i < 4; ++i)
      afr[i] = load_frag(&As[buf][0][0], 40, wm + i * 16, 0, lane);
#pragma unroll
    for (int j = 0; j < 4; ++j) {
      v16bf bfr = load_frag(&Bs[buf][0][0], 40, wn + j * 16, 0, lane);
#pragma unroll
      for (int i = 0; i < 4; ++i)
        acc[i][j] = wmma_bf16(afr[i], bfr, acc[i][j]);
    }
    __syncthreads();                       // done reading before restage
  }

  const int rbase = (lane >> 4) << 3;
  const int cl    = lane & 15;
#pragma unroll
  for (int j = 0; j < 4; ++j) {
    const int col = nblk + wn + j * 16 + cl;
    const float bv = bias[col];
#pragma unroll
    for (int i = 0; i < 4; ++i) {
#pragma unroll
      for (int r = 0; r < 8; ++r) {
        const int row = mblk + wm + i * 16 + rbase + r;
        float v = acc[i][j][r] + bv;
        if (RELU) v = fmaxf(v, 0.0f);
        if (OUT_BF16)
          ((bf16*)Cout)[(size_t)row * N + col] = (bf16)v;
        else
          ((float*)Cout)[(size_t)row * N + col] = v;
      }
    }
  }
}

// ---------------------------------------------------------------------------
// Flash-style MHA for S=512, hd=64, H=8.
// Block = 256 thr (8 waves); each wave owns 16 query rows -> 128 q-rows/block.
// grid = (S/128, H, B). Keys streamed in 32-wide chunks, online softmax.
// K-chunk async-staged to LDS row-major; V-chunk staged transposed so PV
// B-fragments are contiguous ds_load_b128. P goes C-layout -> A-layout via a
// wave-private LDS scratch (wave-internal, no cross-wave sync).
// ---------------------------------------------------------------------------
__global__ __launch_bounds__(256) void k_attn(
    const bf16* __restrict__ qb, int ldq,
    const bf16* __restrict__ kb, const bf16* __restrict__ vb, int ldkv,
    const float* __restrict__ mask, int useclip,
    float* __restrict__ out) {
  __shared__ __align__(16) bf16 Kc[32][72];        // keys x hd (+pad)
  __shared__ __align__(16) bf16 Vt[64][40];        // hd x keys (+pad)
  __shared__ __align__(16) bf16 Pl[8][16 * 32];    // per-wave P scratch

  const int lane  = threadIdx.x & 31;
  const int wid   = threadIdx.x >> 5;
  const int b     = blockIdx.z;
  const int h     = blockIdx.y;
  const int qbase = blockIdx.x * 128 + wid * 16;

  const bf16* q  = qb + (size_t)b * S_ * ldq  + h * HD_;
  const bf16* kp = kb + (size_t)b * S_ * ldkv + h * HD_;
  const bf16* vp = vb + (size_t)b * S_ * ldkv + h * HD_;

  const v16bf qa0 = load_frag(q, ldq, qbase, 0,  lane);
  const v16bf qa1 = load_frag(q, ldq, qbase, 32, lane);

  const int rbase = (lane >> 4) << 3;
  const int cl    = lane & 15;

  float mrow[8], lrow[8];
#pragma unroll
  for (int r = 0; r < 8; ++r) { mrow[r] = -1e30f; lrow[r] = 0.0f; }
  v8f o0 = {}, o1 = {}, o2 = {}, o3 = {};

  const float scale = 0.125f;   // 1/sqrt(64)

  for (int kc = 0; kc < S_; kc += 32) {
    __syncthreads();
    {  // stage K (async, row-major) and V (transposed): 2048 elems each
      const int i   = threadIdx.x * 8;
      const int key = i >> 6;
      const int dd  = i & 63;
      async_copy16(&Kc[key][dd], kp + (size_t)(kc + key) * ldkv + dd);
      v8bf v8v = *(const v8bf*)(vp + (size_t)(kc + key) * ldkv + dd);
#pragma unroll
      for (int e = 0; e < 8; ++e) Vt[dd + e][key] = v8v[e];
    }
    async_wait0();
    __syncthreads();

    // ---- scores: S(16 x 32) = Q(16x64) @ K^T ----
    v8f s0 = {}, s1 = {};
    {
      v16bf kb0 = load_frag(&Kc[0][0], 72, 0,  0, lane);
      v16bf kb1 = load_frag(&Kc[0][0], 72, 16, 0, lane);
      s0 = wmma_bf16(qa0, kb0, s0);
      s1 = wmma_bf16(qa0, kb1, s1);
      kb0 = load_frag(&Kc[0][0], 72, 0,  32, lane);
      kb1 = load_frag(&Kc[0][0], 72, 16, 32, lane);
      s0 = wmma_bf16(qa1, kb0, s0);
      s1 = wmma_bf16(qa1, kb1, s1);
    }

    // ---- scale / clip-tanh / mask ----
#pragma unroll
    for (int r = 0; r < 8; ++r) {
      float a0 = s0[r] * scale;
      float a1 = s1[r] * scale;
      if (useclip) { a0 = 10.0f * tanhf(a0); a1 = 10.0f * tanhf(a1); }
      if (mask) {
        const size_t mb = ((size_t)b * S_ + (qbase + rbase + r)) * S_ + kc;
        a0 += mask[mb + cl];
        a1 += mask[mb + 16 + cl];
      }
      s0[r] = a0;
      s1[r] = a1;
    }

    // ---- online softmax update (row reductions over 16-lane groups) ----
#pragma unroll
    for (int r = 0; r < 8; ++r) {
      float cm = fmaxf(s0[r], s1[r]);
      cm = fmaxf(cm, __shfl_xor(cm, 1));
      cm = fmaxf(cm, __shfl_xor(cm, 2));
      cm = fmaxf(cm, __shfl_xor(cm, 4));
      cm = fmaxf(cm, __shfl_xor(cm, 8));
      const float mn   = fmaxf(mrow[r], cm);
      const float corr = __expf(mrow[r] - mn);
      const float p0   = __expf(s0[r] - mn);
      const float p1   = __expf(s1[r] - mn);
      float rs = p0 + p1;
      rs += __shfl_xor(rs, 1);
      rs += __shfl_xor(rs, 2);
      rs += __shfl_xor(rs, 4);
      rs += __shfl_xor(rs, 8);
      lrow[r] = lrow[r] * corr + rs;
      mrow[r] = mn;
      o0[r] *= corr; o1[r] *= corr; o2[r] *= corr; o3[r] *= corr;
      Pl[wid][(rbase + r) * 32 + cl]      = (bf16)p0;
      Pl[wid][(rbase + r) * 32 + 16 + cl] = (bf16)p1;
    }

    // ---- O += P(16x32) @ V(32x64) ----
    {
      const bf16* pp = &Pl[wid][(lane & 15) * 32 + ((lane >> 4) << 3)];
      v16bf pa;
      *(v8bf*)&pa       = *(const v8bf*)(pp);
      *((v8bf*)&pa + 1) = *(const v8bf*)(pp + 16);

      auto vfrag = [&](int f) {
        const bf16* p = &Vt[f * 16 + cl][(lane >> 4) << 3];
        v16bf v;
        *(v8bf*)&v       = *(const v8bf*)(p);
        *((v8bf*)&v + 1) = *(const v8bf*)(p + 16);
        return v;
      };
      o0 = wmma_bf16(pa, vfrag(0), o0);
      o1 = wmma_bf16(pa, vfrag(1), o1);
      o2 = wmma_bf16(pa, vfrag(2), o2);
      o3 = wmma_bf16(pa, vfrag(3), o3);
    }
  }

  float* ob = out + (size_t)b * S_ * D_ + h * HD_;
#pragma unroll
  for (int r = 0; r < 8; ++r) {
    const float inv = 1.0f / lrow[r];
    float* po = ob + (size_t)(qbase + rbase + r) * D_;
    po[0 * 16 + cl] = o0[r] * inv;
    po[1 * 16 + cl] = o1[r] * inv;
    po[2 * 16 + cl] = o2[r] * inv;
    po[3 * 16 + cl] = o3[r] * inv;
  }
}

// ---------------------------------------------------------------------------
// Fused residual + LayerNorm: out = LN(a + b) * g + be, optional bf16 copy.
// ---------------------------------------------------------------------------
__global__ __launch_bounds__(256) void k_ln_res(
    const float* __restrict__ a, const float* __restrict__ bsrc,
    const float* __restrict__ g, const float* __restrict__ be,
    float* __restrict__ out, bf16* __restrict__ outbf) {
  const int row  = blockIdx.x * 8 + (threadIdx.x >> 5);
  const int lane = threadIdx.x & 31;
  const float* pa = a    + (size_t)row * D_;
  const float* pb = bsrc + (size_t)row * D_;

  float v[16];
  float s = 0.0f;
#pragma unroll
  for (int i = 0; i < 16; ++i) {
    const int c = lane + i * 32;
    v[i] = pa[c] + pb[c];
    s += v[i];
  }
#pragma unroll
  for (int off = 1; off < 32; off <<= 1) s += __shfl_xor(s, off);
  const float mean = s * (1.0f / D_);

  float s2 = 0.0f;
#pragma unroll
  for (int i = 0; i < 16; ++i) {
    const float d = v[i] - mean;
    s2 += d * d;
  }
#pragma unroll
  for (int off = 1; off < 32; off <<= 1) s2 += __shfl_xor(s2, off);
  const float rstd = rsqrtf(s2 * (1.0f / D_) + 1e-5f);

#pragma unroll
  for (int i = 0; i < 16; ++i) {
    const int c = lane + i * 32;
    const float o = (v[i] - mean) * rstd * g[c] + be[c];
    out[(size_t)row * D_ + c] = o;
    if (outbf) outbf[(size_t)row * D_ + c] = (bf16)o;
  }
}

// ---------------------------------------------------------------------------
// Host orchestration
// ---------------------------------------------------------------------------
extern "C" void kernel_launch(void* const* d_in, const int* in_sizes, int n_in,
                              void* d_out, int out_size, void* d_ws,
                              size_t ws_size, hipStream_t stream) {
  (void)in_sizes; (void)n_in; (void)out_size; (void)ws_size;

  const float* x     = (const float*)d_in[0];
  const float* mask  = (const float*)d_in[1];
  const float* qkv_w = (const float*)d_in[2];
  const float* qkv_b = (const float*)d_in[3];
  const float* ln1_g = (const float*)d_in[4];
  const float* ln1_b = (const float*)d_in[5];
  const float* kv_w  = (const float*)d_in[6];
  const float* kv_b  = (const float*)d_in[7];
  const float* ln2_g = (const float*)d_in[8];
  const float* ln2_b = (const float*)d_in[9];
  const float* w1    = (const float*)d_in[10];
  const float* b1    = (const float*)d_in[11];
  const float* w2    = (const float*)d_in[12];
  const float* b2    = (const float*)d_in[13];
  const float* ln3_g = (const float*)d_in[14];
  const float* ln3_b = (const float*)d_in[15];

  char* ws = (char*)d_ws;
  size_t off = 0;
  auto alloc = [&](size_t bytes) -> void* {
    void* p = ws + off;
    off = (off + bytes + 255) & ~(size_t)255;
    return p;
  };

  bf16*  wqkv = (bf16*)alloc((size_t)LYR * 3 * D_ * D_ * 2);
  bf16*  wkv  = (bf16*)alloc((size_t)LYR * 2 * D_ * D_ * 2);
  bf16*  wf1  = (bf16*)alloc((size_t)LYR * DFF_ * D_ * 2);
  bf16*  wf2  = (bf16*)alloc((size_t)LYR * D_ * DFF_ * 2);
  float* Hbuf = (float*)alloc((size_t)M_ * D_ * 4);
  float* X1   = (float*)alloc((size_t)M_ * D_ * 4);
  float* Ybuf = (float*)alloc((size_t)M_ * D_ * 4);
  bf16*  Hbf  = (bf16*)alloc((size_t)M_ * D_ * 2);
  bf16*  X1bf = (bf16*)alloc((size_t)M_ * D_ * 2);
  bf16*  Ybf  = (bf16*)alloc((size_t)M_ * D_ * 2);
  bf16*  Big  = (bf16*)alloc((size_t)M_ * DFF_ * 2);   // QKV (1536) / F1 (2048)
  bf16*  KVbf = (bf16*)alloc((size_t)M_ * 2 * D_ * 2);
  float* Att  = (float*)alloc((size_t)M_ * D_ * 4);    // attn-out / FFN-out

  auto cvt = [&](const float* src, bf16* dst, long n) {
    long blocks = (n / 4 + 255) / 256;
    if (blocks > 32768) blocks = 32768;
    k_f32_to_bf16<<<dim3((unsigned)blocks), 256, 0, stream>>>(src, dst, n);
  };

  // Weights -> bf16 once.
  cvt(qkv_w, wqkv, (long)LYR * 3 * D_ * D_);
  cvt(kv_w,  wkv,  (long)LYR * 2 * D_ * D_);
  cvt(w1,    wf1,  (long)LYR * DFF_ * D_);
  cvt(w2,    wf2,  (long)LYR * D_ * DFF_);

  (void)hipMemcpyAsync(Hbuf, x, (size_t)M_ * D_ * 4, hipMemcpyDeviceToDevice,
                       stream);

  const dim3 ga(S_ / 128, H_, B_);
  for (int l = 0; l < LYR; ++l) {
    cvt(Hbuf, Hbf, (long)M_ * D_);

    // QKV projection -> bf16
    k_gemm<1, 0><<<dim3(3 * D_ / 256, M_ / 128), 256, 0, stream>>>(
        Hbf, wqkv + (size_t)l * 3 * D_ * D_, qkv_b + (size_t)l * 3 * D_,
        Big, M_, 3 * D_, D_);

    // Self-attention (no clip, no mask)
    k_attn<<<ga, 256, 0, stream>>>(Big, 3 * D_, Big + D_, Big + 2 * D_,
                                   3 * D_, nullptr, 0, Att);

    k_ln_res<<<M_ / 8, 256, 0, stream>>>(Hbuf, Att, ln1_g + (size_t)l * D_,
                                         ln1_b + (size_t)l * D_, X1, X1bf);

    // KV projection -> bf16
    k_gemm<1, 0><<<dim3(2 * D_ / 256, M_ / 128), 256, 0, stream>>>(
        Hbf, wkv + (size_t)l * 2 * D_ * D_, kv_b + (size_t)l * 2 * D_,
        KVbf, M_, 2 * D_, D_);

    // Cross-attention (clip=10*tanh, +mask); query = x1 directly
    k_attn<<<ga, 256, 0, stream>>>(X1bf, D_, KVbf, KVbf + D_, 2 * D_,
                                   mask, 1, Att);

    k_ln_res<<<M_ / 8, 256, 0, stream>>>(X1, Att, ln2_g + (size_t)l * D_,
                                         ln2_b + (size_t)l * D_, Ybuf, Ybf);

    // FFN
    k_gemm<1, 1><<<dim3(DFF_ / 256, M_ / 128), 256, 0, stream>>>(
        Ybf, wf1 + (size_t)l * DFF_ * D_, b1 + (size_t)l * DFF_,
        Big, M_, DFF_, D_);
    k_gemm<0, 0><<<dim3(D_ / 256, M_ / 128), 256, 0, stream>>>(
        Big, wf2 + (size_t)l * D_ * DFF_, b2 + (size_t)l * D_,
        Att, M_, D_, DFF_);

    k_ln_res<<<M_ / 8, 256, 0, stream>>>(Ybuf, Att, ln3_g + (size_t)l * D_,
                                         ln3_b + (size_t)l * D_, Hbuf,
                                         (bf16*)nullptr);
  }

  (void)hipMemcpyAsync(d_out, Hbuf, (size_t)M_ * D_ * 4,
                       hipMemcpyDeviceToDevice, stream);
}
